// ItemFactor_15762529976628
// MI455X (gfx1250) — matrix operand order repaired and checked by reading, not compiled, not run on hardware
//
#include <hip/hip_runtime.h>
#include <hip/hip_bf16.h>

typedef __attribute__((ext_vector_type(16))) _Float16 v16h;
typedef __attribute__((ext_vector_type(8)))  _Float16 v8h;
typedef __attribute__((ext_vector_type(8)))  float    v8f;
typedef __attribute__((ext_vector_type(4)))  float    v4f;

#define NNEI   200          // N neighbors per row
#define NTOK   208          // padded to 13 tiles of 16
#define NTILE  13
#define WAVES  8
#define EPSV   1e-10f

// ---- LDS byte offsets (dynamic shared) ----
#define OFF_WG1 0u          // gu_w1  packed B-frags: 128x64 f16 = 16384
#define OFF_WG2 16384u      // gu_w2  packed B-frags:  64x64 f16 =  8192
#define OFF_WA1 24576u      // att_w1 packed B-frags: 128x64 f16 = 16384
#define OFF_FJT 40960u      // f_jt [208][64] f16              = 26624
#define OFF_STG 67584u      // per-wave stage [8][16][64] f16  = 16384
#define OFF_MIU 83968u      // miu [208] f32                   =   832
#define OFF_RED 84800u      // agg partials [4][64] f32        =  1024
#define OFF_AGG 85824u      // agg [64] f32                    =   256
#define OFF_TMP 86080u      // reduction scratch               =    64
#define LDS_BYTES 86144u

__device__ inline v8f wmma16(v16h a, v16h b, v8f c) {
    // v_wmma_f32_16x16x32_f16: (neg_a, A, neg_b, B, c_mod, C, reuse_a, reuse_b)
    return __builtin_amdgcn_wmma_f32_16x16x32_f16(false, a, false, b, (short)0, c, false, false);
}

// Load a B-fragment: weights were pre-packed so each lane's 16 halves are contiguous.
__device__ inline v16h loadB(const _Float16* base, int kc, int ft, int lane) {
    return *(const v16h*)(base + (((kc * 4 + ft) * 32 + lane) << 4));
}

// Build A-fragment (16x32 f16 layout) from an f32 row in global memory, scaled.
// Element e<8  -> src[c+e], e>=8 -> src[c+16+(e-8)]  (caller passes c = 32*chunk' + lo)
__device__ inline v16h packAf32(const float* __restrict__ src, int c, float scale) {
    v4f a0 = *(const v4f*)(src + c);
    v4f a1 = *(const v4f*)(src + c + 4);
    v4f b0 = *(const v4f*)(src + c + 16);
    v4f b1 = *(const v4f*)(src + c + 20);
    v16h r;
#pragma unroll
    for (int i = 0; i < 4; ++i) {
        r[i]      = (_Float16)(a0[i] * scale);
        r[i + 4]  = (_Float16)(a1[i] * scale);
        r[i + 8]  = (_Float16)(b0[i] * scale);
        r[i + 12] = (_Float16)(b1[i] * scale);
    }
    return r;
}

// Build A-fragment from an f16 row in LDS (row pointer + column offset c = 32*kc + lo).
__device__ inline v16h loadA16(const _Float16* rowp, int c) {
    v8h a = *(const v8h*)(rowp + c);
    v8h b = *(const v8h*)(rowp + c + 16);
    v16h r;
#pragma unroll
    for (int i = 0; i < 8; ++i) { r[i] = a[i]; r[i + 8] = b[i]; }
    return r;
}

// Pack weight W (shape [64 out][Kdim in], row-major f32) into B-fragment order f16.
// packed[((kc*4+ft)*32 + lane)*16 + e] = W[ft*16 + lane%16][kc*32 + (lane<16?0:8) + (e&7) + (e>=8?16:0)]
__device__ inline void packWeight(_Float16* dst, const float* __restrict__ W, int Kdim, int tid) {
    int total = Kdim * 64;
    for (int i = tid; i < total; i += 256) {
        int e  = i & 15;
        int l  = (i >> 4) & 31;
        int ft = (i >> 9) & 3;
        int kc = i >> 11;
        int n  = ft * 16 + (l & 15);
        int k  = kc * 32 + ((l < 16) ? 0 : 8) + (e & 7) + ((e & 8) << 1);
        dst[i] = (_Float16)W[n * Kdim + k];
    }
}

__global__ void __launch_bounds__(256)
itemfactor_fused_kernel(const int*   __restrict__ iids,
                        const int*   __restrict__ pad,       // [B][N][2]
                        const float* __restrict__ user_emb,
                        const float* __restrict__ item_emb,
                        const float* __restrict__ rate_emb,
                        const float* __restrict__ gu_w1, const float* __restrict__ gu_b1,
                        const float* __restrict__ gu_w2, const float* __restrict__ gu_b2,
                        const float* __restrict__ att_w1, const float* __restrict__ att_b1,
                        const float* __restrict__ att_w2, const float* __restrict__ att_b2,
                        const float* __restrict__ agg_w, const float* __restrict__ agg_b,
                        float* __restrict__ out)
{
    extern __shared__ char smem[];
    _Float16* sWG1 = (_Float16*)(smem + OFF_WG1);
    _Float16* sWG2 = (_Float16*)(smem + OFF_WG2);
    _Float16* sWA1 = (_Float16*)(smem + OFF_WA1);
    _Float16* sFJT = (_Float16*)(smem + OFF_FJT);
    _Float16* sSTG = (_Float16*)(smem + OFF_STG);
    float*    sMIU = (float*)(smem + OFF_MIU);
    float*    sRED = (float*)(smem + OFF_RED);
    float*    sAGG = (float*)(smem + OFF_AGG);
    float*    sTMP = (float*)(smem + OFF_TMP);

    const int tid = threadIdx.x;
    const int b   = blockIdx.x;

    // ---- stage weights into LDS in B-fragment order (read via L2; tiny) ----
    packWeight(sWG1, gu_w1, 128, tid);
    packWeight(sWG2, gu_w2,  64, tid);
    packWeight(sWA1, att_w1, 128, tid);
    __syncthreads();

    const int lane = tid & 31;
    const int w    = tid >> 5;
    const int m    = lane & 15;
    const int lo   = (lane < 16) ? 0 : 8;    // A-fragment column-group base
    const int hi8  = (lane < 16) ? 0 : 8;    // C-fragment row-group base
    _Float16* stg  = sSTG + w * 16 * 64;

    const int iid = iids[b];
    const float* ip = item_emb + (size_t)iid * 64;

    for (int t = w; t < NTILE; t += WAVES) {
        const int row   = t * 16 + m;              // token this lane gathers for
        const bool valid = (row < NNEI);
        int uid = 0, rid = 0;
        if (valid) {
            const int* pp = pad + (((size_t)b * NNEI + row) << 1);
            uid = pp[0]; rid = pp[1];
        }
        const float maskf = (valid && uid > 0) ? 1.0f : 0.0f;
        const float* up = user_emb + (size_t)uid * 64;
        const float* rp = rate_emb + (size_t)rid * 64;

        // x = [user_emb ; rate_emb], K=128 -> 4 A-fragments
        v16h ax0 = packAf32(up, lo,      1.0f);
        v16h ax1 = packAf32(up, 32 + lo, 1.0f);
        v16h ax2 = packAf32(rp, lo,      1.0f);
        v16h ax3 = packAf32(rp, 32 + lo, 1.0f);

        // ---- GEMM1: h = relu(x @ gu_w1^T + b1), stage as f16 [16][64] ----
#pragma unroll
        for (int ft = 0; ft < 4; ++ft) {
            v8f c = {};
            c = wmma16(ax0, loadB(sWG1, 0, ft, lane), c);
            c = wmma16(ax1, loadB(sWG1, 1, ft, lane), c);
            c = wmma16(ax2, loadB(sWG1, 2, ft, lane), c);
            c = wmma16(ax3, loadB(sWG1, 3, ft, lane), c);
            const float bb = gu_b1[ft * 16 + m];
#pragma unroll
            for (int r = 0; r < 8; ++r)
                stg[(r + hi8) * 64 + ft * 16 + m] = (_Float16)fmaxf(c[r] + bb, 0.0f);
        }

        // relayout h (same-wave LDS ops are in-order; no barrier needed)
        v16h ah0 = loadA16(stg + m * 64, lo);
        v16h ah1 = loadA16(stg + m * 64, 32 + lo);

        // ---- GEMM2: f_jt = h @ gu_w2^T + b2 -> sFJT (f16) ----
#pragma unroll
        for (int ft = 0; ft < 4; ++ft) {
            v8f c = {};
            c = wmma16(ah0, loadB(sWG2, 0, ft, lane), c);
            c = wmma16(ah1, loadB(sWG2, 1, ft, lane), c);
            const float bb = gu_b2[ft * 16 + m];
#pragma unroll
            for (int r = 0; r < 8; ++r)
                sFJT[((size_t)(t * 16 + r + hi8)) * 64 + ft * 16 + m] = (_Float16)(c[r] + bb);
        }

        // ---- GEMM3: ah = relu([f_jt ; mask*item] @ att_w1^T + b1) ----
        const _Float16* frow = sFJT + (size_t)row * 64;   // pad rows exist (NTOK=208)
        v16h aa0 = loadA16(frow, lo);
        v16h aa1 = loadA16(frow, 32 + lo);
        v16h aa2 = packAf32(ip, lo,      maskf);
        v16h aa3 = packAf32(ip, 32 + lo, maskf);
#pragma unroll
        for (int ft = 0; ft < 4; ++ft) {
            v8f c = {};
            c = wmma16(aa0, loadB(sWA1, 0, ft, lane), c);
            c = wmma16(aa1, loadB(sWA1, 1, ft, lane), c);
            c = wmma16(aa2, loadB(sWA1, 2, ft, lane), c);
            c = wmma16(aa3, loadB(sWA1, 3, ft, lane), c);
            const float bb = att_b1[ft * 16 + m];
#pragma unroll
            for (int r = 0; r < 8; ++r)
                stg[(r + hi8) * 64 + ft * 16 + m] = (_Float16)fmaxf(c[r] + bb, 0.0f);
        }

        // ---- attention logit + masked exp ----
        if (lane < 16) {
            float acc = att_b2[0];
            const _Float16* hr = stg + m * 64;
#pragma unroll 8
            for (int dd = 0; dd < 64; ++dd) acc += (float)hr[dd] * att_w2[dd];
            sMIU[t * 16 + m] = __expf(acc) * maskf;   // pad & masked rows -> 0
        }
    }
    __syncthreads();

    // ---- softmax denominator (one wave) ----
    if (w == 0) {
        float s = 0.0f;
        for (int n = lane; n < NNEI; n += 32) s += sMIU[n];
#pragma unroll
        for (int off = 16; off > 0; off >>= 1) s += __shfl_xor(s, off, 32);
        if (lane == 0) sTMP[0] = 1.0f / (s + EPSV);
    }
    __syncthreads();
    const float inv = sTMP[0];

    // ---- agg[d] = inv * sum_n miu[n] * f_jt[n][d] ----
    {
        const int d = tid & 63, q = tid >> 6;
        float s = 0.0f;
        for (int n = q; n < NNEI; n += 4) s += sMIU[n] * (float)sFJT[(size_t)n * 64 + d];
        sRED[q * 64 + d] = s;
    }
    __syncthreads();
    if (tid < 64)
        sAGG[tid] = (sRED[tid] + sRED[64 + tid] + sRED[128 + tid] + sRED[192 + tid]) * inv;
    __syncthreads();

    // ---- z = relu(agg @ agg_w^T + agg_b) ----
    if (tid < 64) {
        float acc = agg_b[tid];
        const float* wr = agg_w + tid * 64;
#pragma unroll 8
        for (int dd = 0; dd < 64; ++dd) acc += sAGG[dd] * wr[dd];
        out[(size_t)b * 64 + tid] = fmaxf(acc, 0.0f);
    }
}

extern "C" void kernel_launch(void* const* d_in, const int* in_sizes, int n_in,
                              void* d_out, int out_size, void* d_ws, size_t ws_size,
                              hipStream_t stream) {
    (void)n_in; (void)d_ws; (void)ws_size; (void)out_size;
    const int*   iids     = (const int*)  d_in[0];
    const int*   pad      = (const int*)  d_in[1];
    const float* user_emb = (const float*)d_in[2];
    const float* item_emb = (const float*)d_in[3];
    const float* rate_emb = (const float*)d_in[4];
    const float* gu_w1    = (const float*)d_in[5];
    const float* gu_b1    = (const float*)d_in[6];
    const float* gu_w2    = (const float*)d_in[7];
    const float* gu_b2    = (const float*)d_in[8];
    const float* att_w1   = (const float*)d_in[9];
    const float* att_b1   = (const float*)d_in[10];
    const float* att_w2   = (const float*)d_in[11];
    const float* att_b2   = (const float*)d_in[12];
    const float* agg_w    = (const float*)d_in[13];
    const float* agg_b    = (const float*)d_in[14];
    float* out = (float*)d_out;

    const int B = in_sizes[0];   // 4096

    // allow > default dynamic LDS (deterministic; not a stream op)
    hipFuncSetAttribute((const void*)itemfactor_fused_kernel,
                        hipFuncAttributeMaxDynamicSharedMemorySize, (int)LDS_BYTES);

    itemfactor_fused_kernel<<<B, 256, LDS_BYTES, stream>>>(
        iids, pad, user_emb, item_emb, rate_emb,
        gu_w1, gu_b1, gu_w2, gu_b2,
        att_w1, att_b1, att_w2, att_b2,
        agg_w, agg_b, out);
}
